// SparseMoE_21835613733543
// MI455X (gfx1250) — compile-verified, matrix operand order checked by reading, not run on hardware
//
#include <hip/hip_runtime.h>

typedef __attribute__((ext_vector_type(16))) __bf16 v16bf;
typedef __attribute__((ext_vector_type(8)))  float  v8f;

#define T_TOK  16384
#define E_EXP  8
#define CAP    5120
#define DM     1024
#define DF     2048
#define NSLOT  (T_TOK * 2)
#define NCHUNK 32

#define BM  128
#define BN  64
#define BK  32
#define LDT 40   // padded LDS row stride in halves (80B -> conflict-free b128 reads)

struct Frag16 { union { v16bf v; uint4 q[2]; }; };

// CDNA5 async global->LDS copy (16B per lane), ASYNCcnt-tracked, no VGPR staging.
// VDST = per-lane LDS byte offset (low 32 bits of generic LDS address),
// VADDR = per-lane 64-bit global address.
__device__ __forceinline__ void async_copy_b128(const __bf16* gsrc, const __bf16* ldst) {
  unsigned l = (unsigned)(unsigned long long)ldst;
  asm volatile("global_load_async_to_lds_b128 %0, %1, off"
               :: "v"(l), "v"(gsrc)
               : "memory");
}

// ---------------- Router: logits, top-2, softmax(top2), aux probs ----------------
__global__ void moe_router_kernel(const float* __restrict__ x, const float* __restrict__ Wg,
                                  int* __restrict__ topk_id, float* __restrict__ topk_w,
                                  float* __restrict__ probs_sum) {
  __shared__ float pacc[E_EXP];
  int tid = threadIdx.x;
  if (tid < E_EXP) pacc[tid] = 0.0f;
  __syncthreads();
  int wid = tid >> 5, lane = tid & 31;
  int t = blockIdx.x * 8 + wid;
  const float* xr = x + (size_t)t * DM;
  float s[E_EXP];
#pragma unroll
  for (int e = 0; e < E_EXP; e++) s[e] = 0.0f;
  for (int i = lane; i < DM; i += 32) {
    float xv = xr[i];
#pragma unroll
    for (int e = 0; e < E_EXP; e++) s[e] += xv * Wg[e * DM + i];
  }
#pragma unroll
  for (int off = 16; off > 0; off >>= 1) {
#pragma unroll
    for (int e = 0; e < E_EXP; e++) s[e] += __shfl_xor(s[e], off, 32);
  }
  if (lane == 0) {
    float v1 = -1e30f, v2 = -1e30f; int i1 = 0, i2 = 0;
#pragma unroll
    for (int e = 0; e < E_EXP; e++) {
      float l = s[e];
      if (l > v1)      { v2 = v1; i2 = i1; v1 = l; i1 = e; }
      else if (l > v2) { v2 = l;  i2 = e; }
    }
    float e2 = __expf(v2 - v1);
    float inv12 = 1.0f / (1.0f + e2);
    topk_id[t * 2 + 0] = i1;  topk_id[t * 2 + 1] = i2;
    topk_w [t * 2 + 0] = inv12;
    topk_w [t * 2 + 1] = e2 * inv12;
    // full softmax for aux loss
    float m = s[0];
#pragma unroll
    for (int e = 1; e < E_EXP; e++) m = fmaxf(m, s[e]);
    float den = 0.0f, pe[E_EXP];
#pragma unroll
    for (int e = 0; e < E_EXP; e++) { pe[e] = __expf(s[e] - m); den += pe[e]; }
    float invd = 1.0f / den;
#pragma unroll
    for (int e = 0; e < E_EXP; e++) atomicAdd(&pacc[e], pe[e] * invd);
  }
  __syncthreads();
  if (tid < E_EXP) atomicAdd(&probs_sum[tid], pacc[tid]);
}

// ---------------- Capacity scan (order-preserving) ----------------
__global__ void moe_count_kernel(const int* __restrict__ topk_id, int* __restrict__ cc) {
  __shared__ int cnt[E_EXP];
  if (threadIdx.x < E_EXP) cnt[threadIdx.x] = 0;
  __syncthreads();
  int s = blockIdx.x * 1024 + threadIdx.x;
  atomicAdd(&cnt[topk_id[s]], 1);
  __syncthreads();
  if (threadIdx.x < E_EXP) cc[blockIdx.x * E_EXP + threadIdx.x] = cnt[threadIdx.x];
}

__global__ void moe_base_kernel(const int* __restrict__ cc, int* __restrict__ cb) {
  int e = threadIdx.x;
  if (e < E_EXP) {
    int run = 0;
    for (int c = 0; c < NCHUNK; c++) { cb[c * E_EXP + e] = run; run += cc[c * E_EXP + e]; }
  }
}

__global__ void moe_pos_kernel(const int* __restrict__ topk_id, const int* __restrict__ cb,
                               int* __restrict__ pos) {
  int lane = threadIdx.x;
  int chunk = blockIdx.x;
  int counters[E_EXP];
#pragma unroll
  for (int e = 0; e < E_EXP; e++) counters[e] = cb[chunk * E_EXP + e];
  unsigned long long lmask = ((unsigned long long)1 << lane) - 1ull;
  for (int g = 0; g < 32; g++) {
    int s = chunk * 1024 + g * 32 + lane;
    int e = topk_id[s];
    int p = 0;
#pragma unroll
    for (int ee = 0; ee < E_EXP; ee++) {
      unsigned long long m = __ballot(e == ee);
      if (e == ee) p = counters[ee] + __popcll(m & lmask);
      counters[ee] += __popcll(m);
    }
    pos[s] = (p < CAP) ? p : -1;
  }
}

// ---------------- Dispatch: gather x rows -> bf16 Xd[E][CAP][DM] ----------------
__global__ void moe_dispatch_kernel(const float* __restrict__ x, const int* __restrict__ topk_id,
                                    const int* __restrict__ pos, __bf16* __restrict__ Xd) {
  int wid = threadIdx.x >> 5, lane = threadIdx.x & 31;
  int s = blockIdx.x * 8 + wid;
  int p = pos[s];
  if (p < 0) return;                        // uniform per wave
  int e = topk_id[s];
  int tok = s >> 1;
  const float4* src = (const float4*)(x + (size_t)tok * DM);
  __bf16* dst = Xd + ((size_t)e * CAP + p) * DM;
#pragma unroll
  for (int i = 0; i < 8; i++) {
    float4 v = src[i * 32 + lane];
    union { __bf16 b[4]; unsigned long long u; } pk;
    pk.b[0] = (__bf16)v.x; pk.b[1] = (__bf16)v.y; pk.b[2] = (__bf16)v.z; pk.b[3] = (__bf16)v.w;
    *(unsigned long long*)(dst + (size_t)(i * 32 + lane) * 4) = pk.u;
  }
}

// ---------------- Transpose + convert fp32 [R][C] -> bf16 [C][R], per expert ----------------
__global__ void moe_transpose_bf16_kernel(const float* __restrict__ src, __bf16* __restrict__ dst,
                                          int R, int C) {
  __shared__ float tile[32][33];
  const float* s = src + (size_t)blockIdx.z * R * C;
  __bf16*      d = dst + (size_t)blockIdx.z * R * C;
  int tx = threadIdx.x, ty = threadIdx.y;
  int c0 = blockIdx.x * 32, r0 = blockIdx.y * 32;
#pragma unroll
  for (int j = 0; j < 4; j++)
    tile[ty + j * 8][tx] = s[(size_t)(r0 + ty + j * 8) * C + c0 + tx];
  __syncthreads();
#pragma unroll
  for (int j = 0; j < 4; j++)
    d[(size_t)(c0 + ty + j * 8) * R + r0 + tx] = (__bf16)tile[tx][ty + j * 8];
}

// ---------------- GEMM1: A2 = bf16( silu(X W1) * (X W3) )  (NT, WMMA bf16) ----------------
// Double-buffered LDS, async global->LDS copies overlapped with WMMA of current tile.
__global__ __launch_bounds__(256) void moe_gemm1_kernel(
    const __bf16* __restrict__ Xd, const __bf16* __restrict__ W1t,
    const __bf16* __restrict__ W3t, __bf16* __restrict__ A2) {
  __shared__ __bf16 As [2 * BM * LDT];
  __shared__ __bf16 B1s[2 * BN * LDT];
  __shared__ __bf16 B3s[2 * BN * LDT];
  const int e = blockIdx.z;
  const __bf16* Ag  = Xd  + (size_t)e * CAP * DM + (size_t)blockIdx.x * BM * DM;
  const __bf16* B1g = W1t + (size_t)e * DF  * DM + (size_t)blockIdx.y * BN * DM;
  const __bf16* B3g = W3t + (size_t)e * DF  * DM + (size_t)blockIdx.y * BN * DM;
  const int tid = threadIdx.x, lane = tid & 31, wid = tid >> 5;
  const int wm = (wid & 3) * 32;
  const int wn = (wid >> 2) * 32;
  const int lr = lane & 15;
  const int kb = (lane < 16) ? 0 : 8;

  // per-thread async-copy addressing (one 16B chunk per lane per instruction)
  const int rowT = tid >> 2;            // 0..63
  const int ccT  = (tid & 3) * 8;       // half offset within 32-half row
  const __bf16* gA0 = Ag  + (size_t)rowT        * DM + ccT;
  const __bf16* gA1 = Ag  + (size_t)(rowT + 64) * DM + ccT;
  const __bf16* gB1 = B1g + (size_t)rowT        * DM + ccT;
  const __bf16* gB3 = B3g + (size_t)rowT        * DM + ccT;
  __bf16* lA0 = As  + rowT * LDT + ccT;
  __bf16* lA1 = As  + (rowT + 64) * LDT + ccT;
  __bf16* lB1 = B1s + rowT * LDT + ccT;
  __bf16* lB3 = B3s + rowT * LDT + ccT;

  const v8f vzero = {0.f, 0.f, 0.f, 0.f, 0.f, 0.f, 0.f, 0.f};
  v8f ach[2][2], acg[2][2];
#pragma unroll
  for (int i = 0; i < 2; i++)
#pragma unroll
    for (int j = 0; j < 2; j++) { ach[i][j] = vzero; acg[i][j] = vzero; }

  auto issue_tile = [&](int k0, int buf) {
    async_copy_b128(gA0 + k0, lA0 + buf * (BM * LDT));
    async_copy_b128(gA1 + k0, lA1 + buf * (BM * LDT));
    async_copy_b128(gB1 + k0, lB1 + buf * (BN * LDT));
    async_copy_b128(gB3 + k0, lB3 + buf * (BN * LDT));
  };

  issue_tile(0, 0);
  for (int k0 = 0; k0 < DM; k0 += BK) {
    const int buf = (k0 >> 5) & 1;
    if (k0 + BK < DM) {
      issue_tile(k0 + BK, buf ^ 1);                       // prefetch next tile into other buffer
      __builtin_prefetch(gA0 + k0 + 2 * BK, 0, 3);        // stream A two tiles ahead
      asm volatile("s_wait_asynccnt 0x4" ::: "memory");   // drain current tile only (in-order)
    } else {
      asm volatile("s_wait_asynccnt 0x0" ::: "memory");
    }
    __syncthreads();    // all waves' current tile resident in LDS

    const __bf16* Ab  = As  + buf * (BM * LDT);
    const __bf16* B1b = B1s + buf * (BN * LDT);
    const __bf16* B3b = B3s + buf * (BN * LDT);
    Frag16 a[2], b1[2], b3[2];
#pragma unroll
    for (int mt = 0; mt < 2; mt++) {
      int r = wm + mt * 16 + lr;
      a[mt].q[0] = *(const uint4*)&Ab[r * LDT + kb];
      a[mt].q[1] = *(const uint4*)&Ab[r * LDT + kb + 16];
    }
#pragma unroll
    for (int nt = 0; nt < 2; nt++) {
      int n = wn + nt * 16 + lr;
      b1[nt].q[0] = *(const uint4*)&B1b[n * LDT + kb];
      b1[nt].q[1] = *(const uint4*)&B1b[n * LDT + kb + 16];
      b3[nt].q[0] = *(const uint4*)&B3b[n * LDT + kb];
      b3[nt].q[1] = *(const uint4*)&B3b[n * LDT + kb + 16];
    }
#pragma unroll
    for (int mt = 0; mt < 2; mt++)
#pragma unroll
      for (int nt = 0; nt < 2; nt++) {
        ach[mt][nt] = __builtin_amdgcn_wmma_f32_16x16x32_bf16(
            false, a[mt].v, false, b1[nt].v, (short)0, ach[mt][nt], false, false);
        acg[mt][nt] = __builtin_amdgcn_wmma_f32_16x16x32_bf16(
            false, a[mt].v, false, b3[nt].v, (short)0, acg[mt][nt], false, false);
      }
    __syncthreads();    // all reads of this buffer done before it is refilled
  }

  __bf16* Cp = A2 + (size_t)e * CAP * DF;
  const int m0 = blockIdx.x * BM + wm + ((lane >= 16) ? 8 : 0);
  const int n0 = blockIdx.y * BN + wn + lr;
#pragma unroll
  for (int mt = 0; mt < 2; mt++)
#pragma unroll
    for (int nt = 0; nt < 2; nt++)
#pragma unroll
      for (int v = 0; v < 8; v++) {
        float h = ach[mt][nt][v];
        float g = acg[mt][nt][v];
        float sv = (h / (1.0f + __expf(-h))) * g;   // silu(h) * g
        Cp[(size_t)(m0 + mt * 16 + v) * DF + (n0 + nt * 16)] = (__bf16)sv;
      }
}

// ---------------- GEMM2: Y = A2 * W2  (NT, WMMA bf16, fp32 out) ----------------
__global__ __launch_bounds__(256) void moe_gemm2_kernel(
    const __bf16* __restrict__ A2, const __bf16* __restrict__ W2t,
    float* __restrict__ Y) {
  __shared__ __bf16 As[2 * BM * LDT];
  __shared__ __bf16 Bs[2 * BN * LDT];
  const int e = blockIdx.z;
  const __bf16* Ag = A2  + (size_t)e * CAP * DF + (size_t)blockIdx.x * BM * DF;
  const __bf16* Bg = W2t + (size_t)e * DM  * DF + (size_t)blockIdx.y * BN * DF;
  const int tid = threadIdx.x, lane = tid & 31, wid = tid >> 5;
  const int wm = (wid & 3) * 32;
  const int wn = (wid >> 2) * 32;
  const int lr = lane & 15;
  const int kb = (lane < 16) ? 0 : 8;

  const int rowT = tid >> 2;
  const int ccT  = (tid & 3) * 8;
  const __bf16* gA0 = Ag + (size_t)rowT        * DF + ccT;
  const __bf16* gA1 = Ag + (size_t)(rowT + 64) * DF + ccT;
  const __bf16* gB  = Bg + (size_t)rowT        * DF + ccT;
  __bf16* lA0 = As + rowT * LDT + ccT;
  __bf16* lA1 = As + (rowT + 64) * LDT + ccT;
  __bf16* lB  = Bs + rowT * LDT + ccT;

  const v8f vzero = {0.f, 0.f, 0.f, 0.f, 0.f, 0.f, 0.f, 0.f};
  v8f acc[2][2];
#pragma unroll
  for (int i = 0; i < 2; i++)
#pragma unroll
    for (int j = 0; j < 2; j++) acc[i][j] = vzero;

  auto issue_tile = [&](int k0, int buf) {
    async_copy_b128(gA0 + k0, lA0 + buf * (BM * LDT));
    async_copy_b128(gA1 + k0, lA1 + buf * (BM * LDT));
    async_copy_b128(gB  + k0, lB  + buf * (BN * LDT));
  };

  issue_tile(0, 0);
  for (int k0 = 0; k0 < DF; k0 += BK) {
    const int buf = (k0 >> 5) & 1;
    if (k0 + BK < DF) {
      issue_tile(k0 + BK, buf ^ 1);
      __builtin_prefetch(gA0 + k0 + 2 * BK, 0, 3);
      asm volatile("s_wait_asynccnt 0x3" ::: "memory");
    } else {
      asm volatile("s_wait_asynccnt 0x0" ::: "memory");
    }
    __syncthreads();

    const __bf16* Ab = As + buf * (BM * LDT);
    const __bf16* Bb = Bs + buf * (BN * LDT);
    Frag16 a[2], b[2];
#pragma unroll
    for (int mt = 0; mt < 2; mt++) {
      int r = wm + mt * 16 + lr;
      a[mt].q[0] = *(const uint4*)&Ab[r * LDT + kb];
      a[mt].q[1] = *(const uint4*)&Ab[r * LDT + kb + 16];
    }
#pragma unroll
    for (int nt = 0; nt < 2; nt++) {
      int n = wn + nt * 16 + lr;
      b[nt].q[0] = *(const uint4*)&Bb[n * LDT + kb];
      b[nt].q[1] = *(const uint4*)&Bb[n * LDT + kb + 16];
    }
#pragma unroll
    for (int mt = 0; mt < 2; mt++)
#pragma unroll
      for (int nt = 0; nt < 2; nt++)
        acc[mt][nt] = __builtin_amdgcn_wmma_f32_16x16x32_bf16(
            false, a[mt].v, false, b[nt].v, (short)0, acc[mt][nt], false, false);
    __syncthreads();
  }

  float* Cp = Y + (size_t)e * CAP * DM;
  const int m0 = blockIdx.x * BM + wm + ((lane >= 16) ? 8 : 0);
  const int n0 = blockIdx.y * BN + wn + lr;
#pragma unroll
  for (int mt = 0; mt < 2; mt++)
#pragma unroll
    for (int nt = 0; nt < 2; nt++)
#pragma unroll
      for (int v = 0; v < 8; v++)
        Cp[(size_t)(m0 + mt * 16 + v) * DM + (n0 + nt * 16)] = acc[mt][nt][v];
}

// ---------------- Combine: out = sum_k w_k * keep_k * Y[e_k][p_k] ----------------
__global__ void moe_combine_kernel(const float* __restrict__ Y, const int* __restrict__ topk_id,
                                   const int* __restrict__ pos, const float* __restrict__ topk_w,
                                   float* __restrict__ out) {
  int wid = threadIdx.x >> 5, lane = threadIdx.x & 31;
  int t = blockIdx.x * 8 + wid;
  int e0 = topk_id[t * 2], e1 = topk_id[t * 2 + 1];
  int p0 = pos[t * 2],     p1 = pos[t * 2 + 1];
  float w0 = topk_w[t * 2], w1 = topk_w[t * 2 + 1];
  if (p0 < 0) { w0 = 0.0f; p0 = 0; }
  if (p1 < 0) { w1 = 0.0f; p1 = 0; }
  const float4* y0 = (const float4*)(Y + ((size_t)e0 * CAP + p0) * DM);
  const float4* y1 = (const float4*)(Y + ((size_t)e1 * CAP + p1) * DM);
  float4* o = (float4*)(out + (size_t)t * DM);
#pragma unroll
  for (int i = 0; i < 8; i++) {
    float4 a = y0[i * 32 + lane], b = y1[i * 32 + lane];
    float4 r;
    r.x = w0 * a.x + w1 * b.x;
    r.y = w0 * a.y + w1 * b.y;
    r.z = w0 * a.z + w1 * b.z;
    r.w = w0 * a.w + w1 * b.w;
    o[i * 32 + lane] = r;
  }
}

__global__ void moe_aux_kernel(const float* __restrict__ probs_sum, float* __restrict__ out_aux) {
  if (threadIdx.x == 0) {
    float s = 0.0f;
    for (int e = 0; e < E_EXP; e++) {
      float pm = probs_sum[e] * (1.0f / (float)T_TOK);
      float d = pm - 1.0f / (float)E_EXP;
      s += d * d;
    }
    out_aux[0] = s;   // mean((p-1/E)^2)*E == sum((p-1/E)^2)
  }
}

extern "C" void kernel_launch(void* const* d_in, const int* in_sizes, int n_in,
                              void* d_out, int out_size, void* d_ws, size_t ws_size,
                              hipStream_t stream) {
  (void)in_sizes; (void)n_in; (void)out_size; (void)ws_size;
  const float* x  = (const float*)d_in[0];
  const float* Wg = (const float*)d_in[1];
  const float* W1 = (const float*)d_in[2];
  const float* W3 = (const float*)d_in[3];
  const float* W2 = (const float*)d_in[4];

  char* ws = (char*)d_ws;
  size_t off = 0;
  auto alloc = [&](size_t bytes) -> void* {
    void* p = ws + off;
    off = (off + bytes + 255) & ~(size_t)255;
    return p;
  };
  int*    topk_id   = (int*)   alloc((size_t)NSLOT * 4);
  float*  topk_w    = (float*) alloc((size_t)NSLOT * 4);
  int*    pos       = (int*)   alloc((size_t)NSLOT * 4);
  int*    cc        = (int*)   alloc((size_t)NCHUNK * E_EXP * 4);
  int*    cb        = (int*)   alloc((size_t)NCHUNK * E_EXP * 4);
  float*  probs_sum = (float*) alloc((size_t)E_EXP * 4);
  __bf16* Xd        = (__bf16*)alloc((size_t)E_EXP * CAP * DM * 2);
  __bf16* W1t       = (__bf16*)alloc((size_t)E_EXP * DM * DF * 2);
  __bf16* W3t       = (__bf16*)alloc((size_t)E_EXP * DM * DF * 2);
  __bf16* W2t       = (__bf16*)alloc((size_t)E_EXP * DF * DM * 2);
  __bf16* A2        = (__bf16*)alloc((size_t)E_EXP * CAP * DF * 2);
  float*  Y         = (float*) alloc((size_t)E_EXP * CAP * DM * 4);

  float* out = (float*)d_out;
  float* out_aux = out + (size_t)T_TOK * DM;

  hipMemsetAsync(probs_sum, 0, (size_t)E_EXP * 4, stream);
  hipMemsetAsync(Xd, 0, (size_t)E_EXP * CAP * DM * 2, stream);

  moe_router_kernel<<<T_TOK / 8, 256, 0, stream>>>(x, Wg, topk_id, topk_w, probs_sum);
  moe_count_kernel<<<NCHUNK, 1024, 0, stream>>>(topk_id, cc);
  moe_base_kernel<<<1, 32, 0, stream>>>(cc, cb);
  moe_pos_kernel<<<NCHUNK, 32, 0, stream>>>(topk_id, cb, pos);
  moe_dispatch_kernel<<<NSLOT / 8, 256, 0, stream>>>(x, topk_id, pos, Xd);

  moe_transpose_bf16_kernel<<<dim3(DF / 32, DM / 32, E_EXP), dim3(32, 8), 0, stream>>>(W1, W1t, DM, DF);
  moe_transpose_bf16_kernel<<<dim3(DF / 32, DM / 32, E_EXP), dim3(32, 8), 0, stream>>>(W3, W3t, DM, DF);
  moe_transpose_bf16_kernel<<<dim3(DM / 32, DF / 32, E_EXP), dim3(32, 8), 0, stream>>>(W2, W2t, DF, DM);

  moe_gemm1_kernel<<<dim3(CAP / BM, DF / BN, E_EXP), 256, 0, stream>>>(Xd, W1t, W3t, A2);
  moe_gemm2_kernel<<<dim3(CAP / BM, DM / BN, E_EXP), 256, 0, stream>>>(A2, W2t, Y);

  moe_combine_kernel<<<T_TOK / 8, 256, 0, stream>>>(Y, topk_id, pos, topk_w, out);
  moe_aux_kernel<<<1, 32, 0, stream>>>(probs_sum, out_aux);
}
